// SimpleGNN_88699664597753
// MI455X (gfx1250) — matrix operand order repaired and checked by reading, not compiled, not run on hardware
//
#include <hip/hip_runtime.h>

#define N_NODES 4096
#define N_EDGES 8192
#define HDIM    128
#define INNF    64
#define NGRAPH  8
#define STEPS   3
#define EHID    32

typedef __bf16 v16bf __attribute__((ext_vector_type(16)));
typedef __bf16 v2bf  __attribute__((ext_vector_type(2)));
typedef float  v8f   __attribute__((ext_vector_type(8)));

// Packed f32x2 -> bf16x2 convert (RNE). Prefer the dedicated packed-convert
// builtin when the toolchain exposes it; otherwise a 2-lane vector build lets
// ISel match v_cvt_pk_bf16_f32 directly.
__device__ __forceinline__ unsigned int pack2bf(float lo, float hi) {
#if defined(__has_builtin) && __has_builtin(__builtin_amdgcn_cvt_pk_bf16_f32)
  v2bf v = __builtin_amdgcn_cvt_pk_bf16_f32(lo, hi);
  return __builtin_bit_cast(unsigned int, v);
#else
  v2bf v = {(__bf16)lo, (__bf16)hi};
  return __builtin_bit_cast(unsigned int, v);
#endif
}

// ---------------------------------------------------------------------------
// LDS fragment layouts are pre-swizzled to the exact WMMA VGPR layout so each
// lane reads its whole fragment as one contiguous 32B vector (2x ds_load_b128).
//   A (16-bit 16x32 MxK): lane=(m&15)+16*((k&15)>=8); elem=(k&7)+8*(k>=16)
//   B (16-bit 32x16 KxN): lane=(n&15)+16*((k&15)>=8); elem=(k&7)+8*(k>=16)
//   C (f32 16x16):        lane=n+16*(m>=8);           elem=m&7
// Each staging thread owns fixed (slot, source-offset) pairs; k is even so a
// k-pair lands in one aligned 32-bit LDS word.
// ---------------------------------------------------------------------------

// ---------------------------------------------------------------------------
// Generic bf16 WMMA GEMM: C[M,N] = epilogue(A[M,K] @ B[K,N]).
// Block = 256 thr (8 waves); block tile 64x128; wave tile 16x64 (1 A frag x
// 4 B frags -> 4 v_wmma per chunk). Software-pipelined: next chunk's global
// loads issue between LDS stores and the barrier, overlapping the WMMAs.
// EPI: 1 = relu(C+bias); 2 = relu(C + agg*inv_deg + bias); 3 = C + bias
// TRANSB: 0 -> B[k*ldb+n]; 1 -> B[n*ldb+k]
// ---------------------------------------------------------------------------
template <int TRANSB, int EPI>
__global__ void __launch_bounds__(256)
gemm_bf16_kernel(const float* __restrict__ A, int lda,
                 const float* __restrict__ B, int ldb,
                 const float* __restrict__ bias,
                 float* __restrict__ C, int ldc, int K,
                 const float* __restrict__ agg,
                 const float* __restrict__ deg) {
  __shared__ unsigned short Asw[4 * 32 * 16];   // 4 M-tiles
  __shared__ unsigned short Bsw[8 * 32 * 16];   // 8 N-tiles
  const int m0 = blockIdx.x * 64;
  const int n0 = blockIdx.y * 128;
  const int tid = threadIdx.x, lane = tid & 31, wid = tid >> 5;
  const int mt = wid & 3;
  const int nt0 = (wid >> 2) << 2;

  // Chunk-invariant staging coordinates.
  int aSlot[4], aOff[4];
#pragma unroll
  for (int i = 0; i < 4; ++i) {
    int e = tid + (i << 8);          // 0..1023
    int m = e >> 4, k = (e & 15) << 1;
    aSlot[i] = (((m >> 4) * 32 + ((m & 15) + (((k & 15) >= 8) ? 16 : 0))) << 4)
               + (k & 7) + ((k >= 16) ? 8 : 0);
    aOff[i] = (m0 + m) * lda + k;
  }
  int bSlot[8], bOff[8];
#pragma unroll
  for (int i = 0; i < 8; ++i) {
    int e = tid + (i << 8);          // 0..2047
    int k = (e >> 7) << 1, n = e & 127;
    bSlot[i] = (((n >> 4) * 32 + ((n & 15) + (((k & 15) >= 8) ? 16 : 0))) << 4)
               + (k & 7) + ((k >= 16) ? 8 : 0);
    bOff[i] = TRANSB ? (n0 + n) * ldb + k : k * ldb + (n0 + n);
  }
  const int bStepK = TRANSB ? 1 : ldb;           // k -> k+1
  const int bAdv = TRANSB ? 32 : 32 * ldb;       // chunk advance

  float2 aCur[4], bCur[8];
#pragma unroll
  for (int i = 0; i < 4; ++i)
    aCur[i] = make_float2(A[aOff[i]], A[aOff[i] + 1]);
#pragma unroll
  for (int i = 0; i < 8; ++i)
    bCur[i] = make_float2(B[bOff[i]], B[bOff[i] + bStepK]);

  v8f c[4] = {{}, {}, {}, {}};
  const int nchunk = K >> 5;
  for (int kc = 0; kc < nchunk; ++kc) {
#pragma unroll
    for (int i = 0; i < 4; ++i)
      *(unsigned int*)&Asw[aSlot[i]] = pack2bf(aCur[i].x, aCur[i].y);
#pragma unroll
    for (int i = 0; i < 8; ++i)
      *(unsigned int*)&Bsw[bSlot[i]] = pack2bf(bCur[i].x, bCur[i].y);
    if (kc + 1 < nchunk) {                        // prefetch next chunk
      int ao = (kc + 1) << 5, bo = bAdv * (kc + 1);
#pragma unroll
      for (int i = 0; i < 4; ++i)
        aCur[i] = make_float2(A[aOff[i] + ao], A[aOff[i] + ao + 1]);
#pragma unroll
      for (int i = 0; i < 8; ++i)
        bCur[i] = make_float2(B[bOff[i] + bo], B[bOff[i] + bo + bStepK]);
    }
    __syncthreads();
    v16bf a = *(const v16bf*)&Asw[(mt * 32 + lane) << 4];
#pragma unroll
    for (int j = 0; j < 4; ++j) {
      v16bf b = *(const v16bf*)&Bsw[((nt0 + j) * 32 + lane) << 4];
      c[j] = __builtin_amdgcn_wmma_f32_16x16x32_bf16(false, a, false, b,
                                                     (short)0, c[j], false,
                                                     false);
    }
    __syncthreads();
  }

  const int ncol = lane & 15;
  const int mofs = (lane >> 4) << 3;
#pragma unroll
  for (int j = 0; j < 4; ++j) {
#pragma unroll
    for (int r = 0; r < 8; ++r) {
      int gm = m0 + mt * 16 + mofs + r;
      int gn = n0 + (nt0 + j) * 16 + ncol;
      float v = c[j][r];
      if (EPI == 1) {
        v = fmaxf(v + bias[gn], 0.0f);
      } else if (EPI == 2) {
        float d = deg[gm];
        float inv = d > 0.0f ? 1.0f / d : 0.0f;
        v = fmaxf(v + agg[gm * HDIM + gn] * inv + bias[gn], 0.0f);
      } else if (EPI == 3) {
        v = v + bias[gn];
      }
      C[gm * ldc + gn] = v;
    }
  }
}

// ---------------------------------------------------------------------------
// Edge message GEMM: tile of 64 edges. A[e, kk] = s_{e,kk>>7} * x[row_e,kk&127]
// built on the fly from LDS (33rd k-group with s=1 folds nn_b2);
// B = [nn_w2[t]; nn_b2[t]] streamed from global (chunks 0..127 are w2, chunks
// 128..131 are b2 — uniform per-chunk pointer select). Result scattered via
// global_atomic_add_f32, degree-normalized later in the m-kernel epilogue.
// ---------------------------------------------------------------------------
__global__ void __launch_bounds__(256)
edge_msg_kernel(const float* __restrict__ h, const int* __restrict__ row,
                const int* __restrict__ col, const float* __restrict__ eattr,
                const float* __restrict__ w1, const float* __restrict__ b1,
                const float* __restrict__ w2, const float* __restrict__ b2,
                float* __restrict__ agg) {
  __shared__ float Xs[64 * HDIM];               // 32 KB gathered source rows
  __shared__ float Ss[64 * 33];                 // edge-MLP scales (+1.0 group)
  __shared__ unsigned short Asw[4 * 32 * 16];
  __shared__ unsigned short Bsw[8 * 32 * 16];
  const int e0 = blockIdx.x * 64;
  const int tid = threadIdx.x, lane = tid & 31, wid = tid >> 5;

#pragma unroll
  for (int i = 0; i < 8; ++i) {                 // gather 64 x-rows (float4)
    int e = tid + (i << 8);                     // 0..2047
    int r = e >> 5, c4 = (e & 31) << 2;
    *(float4*)&Xs[r * HDIM + c4] =
        *(const float4*)&h[row[e0 + r] * HDIM + c4];
  }
  for (int i = tid; i < 64 * 33; i += 256) {    // layer-1 scales
    int r = i / 33, k = i - r * 33;
    Ss[i] = (k < EHID) ? fmaxf(eattr[e0 + r] * w1[k] + b1[k], 0.0f) : 1.0f;
  }

  // Chunk-invariant coordinates.
  int aSlot[4], aM[4], aK[4];
#pragma unroll
  for (int i = 0; i < 4; ++i) {
    int e = tid + (i << 8);
    int m = e >> 4, k = (e & 15) << 1;
    aSlot[i] = (((m >> 4) * 32 + ((m & 15) + (((k & 15) >= 8) ? 16 : 0))) << 4)
               + (k & 7) + ((k >= 16) ? 8 : 0);
    aM[i] = m;
    aK[i] = k;
  }
  int bSlot[8], bOff[8];
#pragma unroll
  for (int i = 0; i < 8; ++i) {
    int e = tid + (i << 8);
    int k = (e >> 7) << 1, n = e & 127;
    bSlot[i] = (((n >> 4) * 32 + ((n & 15) + (((k & 15) >= 8) ? 16 : 0))) << 4)
               + (k & 7) + ((k >= 16) ? 8 : 0);
    bOff[i] = k * HDIM + n;
  }

  const float* bp0 = w2;                         // chunk 0 pointer
  float2 bCur[8];
#pragma unroll
  for (int i = 0; i < 8; ++i)
    bCur[i] = make_float2(bp0[bOff[i]], bp0[bOff[i] + HDIM]);

  __syncthreads();                               // Xs/Ss ready

  const int mt = wid & 3;
  const int nt0 = (wid >> 2) << 2;
  v8f c[4] = {{}, {}, {}, {}};

  for (int kc = 0; kc < 132; ++kc) {             // K = 33*128 = 4224
    const int kk0 = kc << 5;
#pragma unroll
    for (int i = 0; i < 4; ++i) {                // A from LDS (scale * x)
      int kk = kk0 + aK[i];
      float s = Ss[aM[i] * 33 + (kk >> 7)];
      const float* xr = &Xs[aM[i] * HDIM + (kk & 127)];
      *(unsigned int*)&Asw[aSlot[i]] = pack2bf(s * xr[0], s * xr[1]);
    }
#pragma unroll
    for (int i = 0; i < 8; ++i)                  // B (prefetched)
      *(unsigned int*)&Bsw[bSlot[i]] = pack2bf(bCur[i].x, bCur[i].y);
    if (kc + 1 < 132) {                          // prefetch next B chunk
      int kn = kc + 1;
      const float* bp = (kn < 128) ? (w2 + (kn << 5) * HDIM)
                                   : (b2 + ((kn - 128) << 5) * HDIM);
#pragma unroll
      for (int i = 0; i < 8; ++i)
        bCur[i] = make_float2(bp[bOff[i]], bp[bOff[i] + HDIM]);
    }
    __syncthreads();
    v16bf a = *(const v16bf*)&Asw[(mt * 32 + lane) << 4];
#pragma unroll
    for (int j = 0; j < 4; ++j) {
      v16bf b = *(const v16bf*)&Bsw[((nt0 + j) * 32 + lane) << 4];
      c[j] = __builtin_amdgcn_wmma_f32_16x16x32_bf16(false, a, false, b,
                                                     (short)0, c[j], false,
                                                     false);
    }
    __syncthreads();
  }

  const int ncol = lane & 15;
  const int mofs = (lane >> 4) << 3;
  int tgt[8];
#pragma unroll
  for (int r = 0; r < 8; ++r) tgt[r] = col[e0 + mt * 16 + mofs + r];
#pragma unroll
  for (int j = 0; j < 4; ++j) {
#pragma unroll
    for (int r = 0; r < 8; ++r) {
      unsafeAtomicAdd(&agg[tgt[r] * HDIM + (nt0 + j) * 16 + ncol], c[j][r]);
    }
  }
}

// ---------------------------------------------------------------------------
__global__ void __launch_bounds__(256)
deg_kernel(const int* __restrict__ col, float* __restrict__ deg) {
  int e = blockIdx.x * 256 + threadIdx.x;
  if (e < N_EDGES) unsafeAtomicAdd(&deg[col[e]], 1.0f);
}

__global__ void __launch_bounds__(256)
gru_gate_kernel(const float* __restrict__ gi, const float* __restrict__ gh,
                float* __restrict__ h) {
  int idx = blockIdx.x * 256 + threadIdx.x;       // N_NODES*HDIM threads
  int n = idx >> 7, o = idx & 127;
  const float* gin = gi + n * 384;
  const float* ghn = gh + n * 384;
  float r = 1.0f / (1.0f + __expf(-(gin[o] + ghn[o])));
  float z = 1.0f / (1.0f + __expf(-(gin[128 + o] + ghn[128 + o])));
  float nn = tanhf(gin[256 + o] + r * ghn[256 + o]);
  float hv = h[idx];
  h[idx] = (1.0f - z) * nn + z * hv;
}

__global__ void __launch_bounds__(256)
pool_kernel(const float* __restrict__ h, const int* __restrict__ batch,
            float* __restrict__ pool, float* __restrict__ cnt) {
  int idx = blockIdx.x * 256 + threadIdx.x;       // N_NODES*HDIM threads
  int n = idx >> 7, o = idx & 127;
  int g = batch[n];
  unsafeAtomicAdd(&pool[g * HDIM + o], h[idx]);
  if (o == 0) unsafeAtomicAdd(&cnt[g], 1.0f);
}

__global__ void __launch_bounds__(512)
readout_kernel(const float* __restrict__ pool, const float* __restrict__ cnt,
               const float* __restrict__ w1, const float* __restrict__ b1,
               const float* __restrict__ w2, const float* __restrict__ b2,
               float* __restrict__ out) {
  __shared__ float g1s[NGRAPH * 64];
  int tid = threadIdx.x;
  int g = tid >> 6, o = tid & 63;                  // 8 graphs x 64 hidden
  float cm = fmaxf(cnt[g], 1.0f);
  float acc = b1[o];
  for (int k = 0; k < HDIM; ++k)
    acc += (pool[g * HDIM + k] / cm) * w1[k * 64 + o];
  g1s[g * 64 + o] = fmaxf(acc, 0.0f);
  __syncthreads();
  if (tid < NGRAPH) {
    float s = b2[0];
    for (int j = 0; j < 64; ++j) s += g1s[tid * 64 + j] * w2[j];
    out[tid] = s;
  }
}

// ---------------------------------------------------------------------------
extern "C" void kernel_launch(void* const* d_in, const int* in_sizes, int n_in,
                              void* d_out, int out_size, void* d_ws,
                              size_t ws_size, hipStream_t stream) {
  const float* x       = (const float*)d_in[0];
  const int*   eidx    = (const int*)d_in[1];
  const float* eattr   = (const float*)d_in[2];
  const int*   batch   = (const int*)d_in[3];
  const float* lin0_w  = (const float*)d_in[4];
  const float* lin0_b  = (const float*)d_in[5];
  const float* nn_w1   = (const float*)d_in[6];
  const float* nn_b1   = (const float*)d_in[7];
  const float* nn_w2   = (const float*)d_in[8];
  const float* nn_b2   = (const float*)d_in[9];
  const float* root_w  = (const float*)d_in[10];
  const float* conv_b  = (const float*)d_in[11];
  const float* gru_wih = (const float*)d_in[12];
  const float* gru_whh = (const float*)d_in[13];
  const float* gru_bih = (const float*)d_in[14];
  const float* gru_bhh = (const float*)d_in[15];
  const float* lin1_w  = (const float*)d_in[16];
  const float* lin1_b  = (const float*)d_in[17];
  const float* lin2_w  = (const float*)d_in[18];
  const float* lin2_b  = (const float*)d_in[19];
  float* out = (float*)d_out;

  const int* row = eidx;
  const int* col = eidx + N_EDGES;

  float* ws   = (float*)d_ws;
  float* hbuf = ws; ws += N_NODES * HDIM;
  float* agg  = ws; ws += N_NODES * HDIM;
  float* mbuf = ws; ws += N_NODES * HDIM;
  float* gi   = ws; ws += N_NODES * 3 * HDIM;
  float* gh   = ws; ws += N_NODES * 3 * HDIM;
  float* deg  = ws; ws += N_NODES;
  float* pool = ws; ws += NGRAPH * HDIM;
  float* cnt  = ws; ws += NGRAPH;

  hipMemsetAsync(deg, 0, N_NODES * sizeof(float), stream);
  deg_kernel<<<N_EDGES / 256, 256, 0, stream>>>(col, deg);

  // lin0: h = relu(x @ lin0_w + lin0_b)
  gemm_bf16_kernel<0, 1><<<dim3(N_NODES / 64, 1), 256, 0, stream>>>(
      x, INNF, lin0_w, HDIM, lin0_b, hbuf, HDIM, INNF, nullptr, nullptr);

  for (int t = 0; t < STEPS; ++t) {
    hipMemsetAsync(agg, 0, N_NODES * HDIM * sizeof(float), stream);
    edge_msg_kernel<<<N_EDGES / 64, 256, 0, stream>>>(
        hbuf, row, col, eattr, nn_w1 + t * EHID, nn_b1 + t * EHID,
        nn_w2 + t * EHID * HDIM * HDIM, nn_b2 + t * HDIM * HDIM, agg);
    // m = relu(agg*inv_deg + h @ root_w[t] + conv_b[t])
    gemm_bf16_kernel<0, 2><<<dim3(N_NODES / 64, 1), 256, 0, stream>>>(
        hbuf, HDIM, root_w + t * HDIM * HDIM, HDIM, conv_b + t * HDIM, mbuf,
        HDIM, HDIM, agg, deg);
    // gi = m @ wih^T + bih ; gh = h @ whh^T + bhh
    gemm_bf16_kernel<1, 3><<<dim3(N_NODES / 64, 3), 256, 0, stream>>>(
        mbuf, HDIM, gru_wih, HDIM, gru_bih, gi, 3 * HDIM, HDIM, nullptr,
        nullptr);
    gemm_bf16_kernel<1, 3><<<dim3(N_NODES / 64, 3), 256, 0, stream>>>(
        hbuf, HDIM, gru_whh, HDIM, gru_bhh, gh, 3 * HDIM, HDIM, nullptr,
        nullptr);
    gru_gate_kernel<<<N_NODES * HDIM / 256, 256, 0, stream>>>(gi, gh, hbuf);
  }

  hipMemsetAsync(pool, 0, (NGRAPH * HDIM + NGRAPH) * sizeof(float), stream);
  pool_kernel<<<N_NODES * HDIM / 256, 256, 0, stream>>>(hbuf, batch, pool, cnt);
  readout_kernel<<<1, 512, 0, stream>>>(pool, cnt, lin1_w, lin1_b, lin2_w,
                                        lin2_b, out);
}
